// Memory_Attention_Layer_47364899340847
// MI455X (gfx1250) — compile-verified, matrix operand order checked by reading, not run on hardware
//
#include <hip/hip_runtime.h>
#include <hip/hip_bf16.h>

// ---------------------------------------------------------------------------
// Memory-Attention layer for MI455X (gfx1250, wave32, WMMA, big-LDS, async).
// One workgroup (8 waves, 256 threads) per batch element.
// - Scores (256x256 f16 = 128 KB) live entirely in LDS (WGP has 320 KB).
// - All GEMMs via v_wmma_f32_16x16x32_f16.
// - Raw k/q/v intermediates are kept as f16 in scratch so K-staging can use
//   the CDNA5 async DMA path: global_load_async_to_lds_b128 + s_wait_asynccnt.
// Total static LDS = 160 KB.
// ---------------------------------------------------------------------------

constexpr int LQ  = 256;   // sequence length
constexpr int CD  = 64;    // channels (C == HID == 64)
constexpr int BB  = 32;    // batch
constexpr int NUM = 64;    // scan steps

typedef __attribute__((ext_vector_type(16))) _Float16 v16h;
typedef __attribute__((ext_vector_type(8)))  _Float16 v8h;
typedef __attribute__((ext_vector_type(4)))  _Float16 v4h;
typedef __attribute__((ext_vector_type(8)))  float    v8f;
typedef __attribute__((ext_vector_type(4)))  float    v4f;

// Load one 16x32 f16 WMMA operand fragment from a row-major [rows][ld] f16
// array (LDS or global).  A-operand: rows are M.  B-operand: pass B^T (rows
// are N) -> identical layout.  Per ISA 7.12.2: lane m=lane&15, hi=lane>>4;
// regs 0..3 hold K=k0+hi*8.., regs 4..7 hold K=k0+16+hi*8.. =>
// two contiguous 16-byte loads (ds_load_b128 / global_load_b128).
__device__ __forceinline__ v16h frag_ld(const _Float16* __restrict__ src,
                                        int ld, int r0, int k0) {
  const int lane = threadIdx.x & 31;
  const int m    = lane & 15;
  const int hi   = lane >> 4;
  const _Float16* p = src + (size_t)(r0 + m) * ld + k0 + hi * 8;
  v8h lo = *(const v8h*)(p);
  v8h hc = *(const v8h*)(p + 16);
  v16h f;
#pragma unroll
  for (int i = 0; i < 8; ++i) { f[i] = lo[i]; f[i + 8] = hc[i]; }
  return f;
}

// Same fragment, but sourced from a row-major f32 global array with
// in-register conversion (used when the operand comes from the f32 carry).
__device__ __forceinline__ v16h frag_ld_gf32(const float* __restrict__ src,
                                             int ld, int r0, int k0) {
  const int lane = threadIdx.x & 31;
  const int m    = lane & 15;
  const int hi   = lane >> 4;
  const float* p = src + (size_t)(r0 + m) * ld + k0 + hi * 8;
  v4f f0 = *(const v4f*)(p);
  v4f f1 = *(const v4f*)(p + 4);
  v4f f2 = *(const v4f*)(p + 16);
  v4f f3 = *(const v4f*)(p + 20);
  v16h f;
#pragma unroll
  for (int i = 0; i < 4; ++i) {
    f[i]      = (_Float16)f0[i];
    f[4 + i]  = (_Float16)f1[i];
    f[8 + i]  = (_Float16)f2[i];
    f[12 + i] = (_Float16)f3[i];
  }
  return f;
}

__device__ __forceinline__ v8f wmma16(v16h a, v16h b, v8f c) {
  // D = A(16x32) * B(32x16) + C, f32 accum.
  return __builtin_amdgcn_wmma_f32_16x16x32_f16(false, a, false, b,
                                                (short)0, c, false, false);
}

// ---------------------------------------------------------------------------
// Per-head K/Q/V projection (8x8 GEMMs -> plain VALU, one thread per output).
// T = float (t=0, straight to f32 outputs) or _Float16 (raw scratch).
// ---------------------------------------------------------------------------
template <typename T>
__global__ __launch_bounds__(256)
void kqv_kernel(const float* __restrict__ xt, long x_bs,
                const float* __restrict__ emb,
                const float* __restrict__ Wk, const float* __restrict__ bk,
                const float* __restrict__ Wq, const float* __restrict__ bq,
                const float* __restrict__ Wv, const float* __restrict__ bv,
                T* __restrict__ ko, T* __restrict__ qo, T* __restrict__ vo,
                long o_bs) {
  const int gid = blockIdx.x * 256 + threadIdx.x;  // [0, B*L*64)
  const int he  = gid & 63;
  const int l   = (gid >> 6) & (LQ - 1);
  const int b   = gid >> 14;
  const int h   = he >> 3, e = he & 7;
  const float* xr = xt + (size_t)b * x_bs + (size_t)l * CD + h * 8;
  const float* er = emb + (size_t)l * CD + h * 8;
  float xv[8];
#pragma unroll
  for (int d = 0; d < 8; ++d) xv[d] = xr[d] + er[d];
  float sk = bk[h * 8 + e], sq = bq[h * 8 + e], sv = bv[h * 8 + e];
#pragma unroll
  for (int d = 0; d < 8; ++d) {
    const int wi = (h * 8 + d) * 8 + e;
    sk += xv[d] * Wk[wi];
    sq += xv[d] * Wq[wi];
    sv += xv[d] * Wv[wi];
  }
  const size_t off = (size_t)b * o_bs + (size_t)l * CD + he;
  ko[off] = (T)sk; qo[off] = (T)sq; vo[off] = (T)sv;
}

// ---------------------------------------------------------------------------
// Fused cross-attention, all-LDS:
//   S = softmax(q k^T) / 8 ; out = S v ; z = LN(out+v) ;
//   y = LN( ReLU(z W1 + b1) W2 + b2 + z )
// KT/QT in {float (carry), _Float16 (raw scratch)}; v is always f16 raw.
// f16 K-staging uses CDNA5 async DMA (global_load_async_to_lds_b128).
// LDS: sS 128 KB (scores; each wave's own rows are recycled to hold z),
//      sB  32 KB (K -> V^T -> W1t|W2t|per-wave scratch).
// ---------------------------------------------------------------------------
template <typename KT, typename QT>
__global__ __launch_bounds__(256)
void attn_kernel(const KT* __restrict__ kin, long kin_bs,
                 const QT* __restrict__ qin, long qin_bs,
                 const _Float16* __restrict__ vin, long vin_bs,
                 const float* __restrict__ g0, const float* __restrict__ b0,
                 const float* __restrict__ g1, const float* __restrict__ b1,
                 const float* __restrict__ w1, const float* __restrict__ bb1,
                 const float* __restrict__ w2, const float* __restrict__ bb2,
                 float* __restrict__ out, long out_bs) {
  __shared__ __align__(16) _Float16 sS[LQ * LQ];  // 128 KB scores / z
  __shared__ __align__(16) _Float16 sB[LQ * CD];  //  32 KB K -> V^T -> weights

  const int tid  = threadIdx.x;
  const int wave = tid >> 5;
  const int lane = tid & 31;
  const int b    = blockIdx.x;

  const KT*       kp = kin + (size_t)b * kin_bs;
  const QT*       qp = qin + (size_t)b * qin_bs;
  const _Float16* vp = vin + (size_t)b * vin_bs;
  float*          op = out + (size_t)b * out_bs;

  __builtin_prefetch(qp, 0, 1);          // global_prefetch_b8
  __builtin_prefetch(vp, 0, 1);
  __builtin_prefetch(w1, 0, 1);
  __builtin_prefetch(w2, 0, 1);

  // --- P0: stage K -> sB --------------------------------------------------
  if constexpr (__is_same(KT, _Float16)) {
    // Pure byte copy: CDNA5 Tensor/async path, LDS dest addr in VDST VGPR,
    // 64-bit global addr pair, GV mode.  Tracked by ASYNCcnt.
    for (int e = tid * 8; e < LQ * CD; e += 2048) {
      const unsigned lds = (unsigned)(uintptr_t)(sB + e);
      const unsigned long long ga = (unsigned long long)(uintptr_t)(kp + e);
      asm volatile("global_load_async_to_lds_b128 %0, %1, off"
                   :: "v"(lds), "v"(ga) : "memory");
    }
    asm volatile("s_wait_asynccnt 0" ::: "memory");
  } else {
    for (int e = tid * 4; e < LQ * CD; e += 1024) {
      v4f f = *(const v4f*)(kp + e);
      v4h h;
#pragma unroll
      for (int i = 0; i < 4; ++i) h[i] = (_Float16)f[i];
      *(v4h*)(sB + e) = h;
    }
  }
  __syncthreads();

  // --- P1: scores S = Q K^T -> sS (unnormalized f16) -----------------------
  // Wave w owns rows [32w, 32w+32) = two 16-row tiles; 16 col tiles; K=64.
#pragma unroll
  for (int rt = 0; rt < 2; ++rt) {
    const int m0 = (wave * 2 + rt) * 16;
    v16h a0, a1;
    if constexpr (__is_same(QT, _Float16)) {
      a0 = frag_ld(qp, CD, m0, 0);
      a1 = frag_ld(qp, CD, m0, 32);
    } else {
      a0 = frag_ld_gf32(qp, CD, m0, 0);
      a1 = frag_ld_gf32(qp, CD, m0, 32);
    }
    const int nl = lane & 15, mh = (lane >> 4) * 8;
    for (int nt = 0; nt < 16; ++nt) {
      const int n0 = nt * 16;
      v8f acc = {};
      acc = wmma16(a0, frag_ld(sB, CD, n0, 0), acc);
      acc = wmma16(a1, frag_ld(sB, CD, n0, 32), acc);
      const int n = n0 + nl;
#pragma unroll
      for (int j = 0; j < 8; ++j)
        sS[(m0 + mh + j) * LQ + n] = (_Float16)acc[j];   // ds_store_b16
    }
  }
  __syncthreads();

  // --- P2: row softmax (max-subtracted), then * 1/sqrt(64), v8h chunks -----
  {
    _Float16* row = sS + tid * LQ;
    float mx = -3.0e38f;
    for (int i = 0; i < LQ; i += 8) {
      v8h c = *(const v8h*)(row + i);
#pragma unroll
      for (int j = 0; j < 8; ++j) {
        const float s = (float)c[j];
        mx = s > mx ? s : mx;
      }
    }
    float sum = 0.f;
    for (int i = 0; i < LQ; i += 8) {
      v8h c = *(const v8h*)(row + i);
#pragma unroll
      for (int j = 0; j < 8; ++j) {
        const float y = __expf((float)c[j] - mx);
        sum += y;
        c[j] = (_Float16)y;
      }
      *(v8h*)(row + i) = c;
    }
    const float inv = 0.125f / sum;
    for (int i = 0; i < LQ; i += 8) {
      v8h c = *(const v8h*)(row + i);
#pragma unroll
      for (int j = 0; j < 8; ++j) c[j] = (_Float16)((float)c[j] * inv);
      *(v8h*)(row + i) = c;
    }
  }
  __syncthreads();

  // --- P3: stage V^T -> sB; out = S V; z = LN(out+v) -> recycled sS rows ---
  for (int e = tid * 4; e < LQ * CD; e += 1024) {
    const int l = e >> 6, c = e & 63;
    v4h f = *(const v4h*)(vp + e);
#pragma unroll
    for (int i = 0; i < 4; ++i) sB[(c + i) * LQ + l] = f[i];
  }
  __syncthreads();

#pragma unroll
  for (int rt = 0; rt < 2; ++rt) {
    const int m0 = (wave * 2 + rt) * 16;
    v8f acc[4] = {{}, {}, {}, {}};
    for (int kt = 0; kt < 8; ++kt) {             // K = 256
      const v16h a = frag_ld(sS, LQ, m0, kt * 32);
#pragma unroll
      for (int ct = 0; ct < 4; ++ct)
        acc[ct] = wmma16(a, frag_ld(sB, LQ, ct * 16, kt * 32), acc[ct]);
    }
    // This wave's S rows m0..m0+15 are now fully consumed -> recycle them
    // (stride LQ) to hold (out + v), then z after in-place LayerNorm.
    const int nl = lane & 15, mh = (lane >> 4) * 8;
#pragma unroll
    for (int ct = 0; ct < 4; ++ct) {
      const int n = ct * 16 + nl;
#pragma unroll
      for (int j = 0; j < 8; ++j) {
        const int m = m0 + mh + j;
        sS[m * LQ + n] =
            (_Float16)(acc[ct][j] + (float)vp[(size_t)m * CD + n]);
      }
    }
    // Wave-local LayerNorm (DS ops in-order within a wave).
    if (lane < 16) {
      const int m = m0 + lane;
      float mean = 0.f;
      for (int c = 0; c < CD; ++c) mean += (float)sS[m * LQ + c];
      mean *= (1.f / CD);
      float var = 0.f;
      for (int c = 0; c < CD; ++c) {
        const float d = (float)sS[m * LQ + c] - mean;
        var += d * d;
      }
      var *= (1.f / CD);
      const float rs = rsqrtf(var + 1e-5f);
      for (int c = 0; c < CD; ++c)
        sS[m * LQ + c] =
            (_Float16)(((float)sS[m * LQ + c] - mean) * rs * g0[c] + b0[c]);
    }
  }
  __syncthreads();

  // --- P4: stage W1^T, W2^T into sB front; back half = per-wave scratch ----
  _Float16* sW1  = sB;                    // [64][64], sW1[n*64+k] = w1[k][n]
  _Float16* sW2  = sB + 4096;
  _Float16* swsc = sB + 8192;             // 8 waves * 16*64 halves
  for (int e = tid; e < 4096; e += 256) {
    const int n = e >> 6, k = e & 63;
    sW1[e] = (_Float16)w1[k * CD + n];
    sW2[e] = (_Float16)w2[k * CD + n];
  }
  __syncthreads();

  // --- P5: FFN (WMMA x4 + ReLU) fused with residual + final LN -------------
  _Float16* ws = swsc + wave * 16 * CD;
#pragma unroll
  for (int rt = 0; rt < 2; ++rt) {
    const int m0 = (wave * 2 + rt) * 16;
    const v16h a0 = frag_ld(sS, LQ, m0, 0);    // z lives at stride LQ
    const v16h a1 = frag_ld(sS, LQ, m0, 32);
    const int nl = lane & 15, mh = (lane >> 4) * 8;
#pragma unroll
    for (int ct = 0; ct < 4; ++ct) {            // h = ReLU(z W1 + b1)
      v8f acc = {};
      acc = wmma16(a0, frag_ld(sW1, CD, ct * 16, 0), acc);
      acc = wmma16(a1, frag_ld(sW1, CD, ct * 16, 32), acc);
      const int n = ct * 16 + nl;
      const float bias = bb1[n];
#pragma unroll
      for (int j = 0; j < 8; ++j) {
        const float h = acc[j] + bias;
        ws[(mh + j) * CD + n] = (_Float16)(h > 0.f ? h : 0.f);
      }
    }
    const v16h h0 = frag_ld(ws, CD, 0, 0);      // read h before overwrite
    const v16h h1 = frag_ld(ws, CD, 0, 32);
#pragma unroll
    for (int ct = 0; ct < 4; ++ct) {            // f = h W2 + b2 + z
      v8f acc = {};
      acc = wmma16(h0, frag_ld(sW2, CD, ct * 16, 0), acc);
      acc = wmma16(h1, frag_ld(sW2, CD, ct * 16, 32), acc);
      const int n = ct * 16 + nl;
      const float bias = bb2[n];
#pragma unroll
      for (int j = 0; j < 8; ++j) {
        const int m = m0 + mh + j;
        const float f = acc[j] + bias + (float)sS[m * LQ + n];  // + z
        ws[(mh + j) * CD + n] = (_Float16)f;
      }
    }
    if (lane < 16) {                            // final LN, float4 stores
      const int m = m0 + lane;
      float mean = 0.f;
      for (int c = 0; c < CD; ++c) mean += (float)ws[lane * CD + c];
      mean *= (1.f / CD);
      float var = 0.f;
      for (int c = 0; c < CD; ++c) {
        const float d = (float)ws[lane * CD + c] - mean;
        var += d * d;
      }
      var *= (1.f / CD);
      const float rs = rsqrtf(var + 1e-5f);
      for (int c = 0; c < CD; c += 4) {
        v4f o;
#pragma unroll
        for (int i = 0; i < 4; ++i)
          o[i] = ((float)ws[lane * CD + c + i] - mean) * rs * g1[c + i] +
                 b1[c + i];
        *(v4f*)(op + (size_t)m * CD + c) = o;   // global_store_b128
      }
    }
  }
}

// ---------------------------------------------------------------------------
// Host driver: 1 + 63*4 = 253 launches, sequential on `stream` (graph-safe).
// d_ws: kraw/qraw/vraw as f16, 1 MB each.
// ---------------------------------------------------------------------------
extern "C" void kernel_launch(void* const* d_in, const int* in_sizes, int n_in,
                              void* d_out, int out_size, void* d_ws,
                              size_t ws_size, hipStream_t stream) {
  (void)in_sizes; (void)n_in; (void)out_size; (void)ws_size;
  const float* x   = (const float*)d_in[0];
  const float* emb = (const float*)d_in[1];
  const float* Wk  = (const float*)d_in[2];  const float* bk = (const float*)d_in[3];
  const float* Wq  = (const float*)d_in[4];  const float* bq = (const float*)d_in[5];
  const float* Wv  = (const float*)d_in[6];  const float* bv = (const float*)d_in[7];
  const float* g0x = (const float*)d_in[8];  const float* b0x = (const float*)d_in[9];
  const float* g1x = (const float*)d_in[10]; const float* b1x = (const float*)d_in[11];
  const float* g0k = (const float*)d_in[12]; const float* b0k = (const float*)d_in[13];
  const float* g1k = (const float*)d_in[14]; const float* b1k = (const float*)d_in[15];
  const float* g0q = (const float*)d_in[16]; const float* b0q = (const float*)d_in[17];
  const float* g1q = (const float*)d_in[18]; const float* b1q = (const float*)d_in[19];
  const float* fx1w = (const float*)d_in[20]; const float* fx1b = (const float*)d_in[21];
  const float* fx2w = (const float*)d_in[22]; const float* fx2b = (const float*)d_in[23];
  const float* fk1w = (const float*)d_in[24]; const float* fk1b = (const float*)d_in[25];
  const float* fk2w = (const float*)d_in[26]; const float* fk2b = (const float*)d_in[27];
  const float* fq1w = (const float*)d_in[28]; const float* fq1b = (const float*)d_in[29];
  const float* fq2w = (const float*)d_in[30]; const float* fq2b = (const float*)d_in[31];

  const size_t TOT = (size_t)BB * NUM * LQ * CD;
  float* K = (float*)d_out;
  float* Q = K + TOT;
  float* V = Q + TOT;

  _Float16* kraw = (_Float16*)d_ws;
  _Float16* qraw = kraw + (size_t)BB * LQ * CD;
  _Float16* vraw = qraw + (size_t)BB * LQ * CD;

  const long xbs = (long)NUM * LQ * CD;   // batch stride in x / outputs
  const long obs = (long)NUM * LQ * CD;
  const long rbs = (long)LQ * CD;         // batch stride in raw buffers
  const int  kqvGrid = (BB * LQ * CD) / 256;

  // t = 0: raw k/q/v go straight to outputs (they are also the first carry).
  kqv_kernel<float><<<kqvGrid, 256, 0, stream>>>(x, xbs, emb, Wk, bk, Wq, bq,
                                                 Wv, bv, K, Q, V, obs);
  for (int t = 1; t < NUM; ++t) {
    const size_t st = (size_t)t * LQ * CD;
    const size_t sp = (size_t)(t - 1) * LQ * CD;
    kqv_kernel<_Float16><<<kqvGrid, 256, 0, stream>>>(
        x + st, xbs, emb, Wk, bk, Wq, bq, Wv, bv, kraw, qraw, vraw, rbs);
    // k_t = cross_attn(k=kraw(f16, async-DMA), q=Q[t-1](f32), v=kraw)
    attn_kernel<_Float16, float><<<BB, 256, 0, stream>>>(
        kraw, rbs, Q + sp, obs, kraw, rbs,
        g0k, b0k, g1k, b1k, fk1w, fk1b, fk2w, fk2b, K + st, obs);
    // q_t = cross_attn(k=K[t-1](f32), q=qraw(f16), v=qraw)
    attn_kernel<float, _Float16><<<BB, 256, 0, stream>>>(
        K + sp, obs, qraw, rbs, qraw, rbs,
        g0q, b0q, g1q, b1q, fq1w, fq1b, fq2w, fq2b, Q + st, obs);
    // v_t = cross_attn(k=K[t](f32), q=Q[t](f32), v=vraw)
    attn_kernel<float, float><<<BB, 256, 0, stream>>>(
        K + st, obs, Q + st, obs, vraw, rbs,
        g0x, b0x, g1x, b1x, fx1w, fx1b, fx2w, fx2b, V + st, obs);
  }
}
